// MambaSSM_41248865911020
// MI455X (gfx1250) — compile-verified, matrix operand order checked by reading, not run on hardware
//
#include <hip/hip_runtime.h>

// MI455X / gfx1250, wave32. Matrix ops via v_wmma_f32_16x16x32_bf16.
//
// Phase P : swizzle+convert weights fp32 -> bf16 WMMA B-fragment layout.
// Phase G1: fused projection GEMM  u = x@Wi.T+bi (bf16 out), sp = x@Ws.T+bs (f32 out).
// Phase S : sequential scan, 1 WG x 8 waves, state exchanged via LDS ping-pong,
//           weights A/B/C resident in VGPR fragments, 12 WMMA / step.
// Phase G2: out = y@Wo.T + bo  (bf16 WMMA, f32 accumulate/store).

#define D_MODEL 1024
#define D_STATE 128
#define BATCH   8
#define SEQ     4096
#define M_TOTAL (BATCH * SEQ)  // 32768

typedef __attribute__((ext_vector_type(16))) __bf16 bf16x16;
typedef __attribute__((ext_vector_type(8)))  float  f32x8;

static __device__ __forceinline__ __bf16 f2bf(float f) {
    union { float f; unsigned u; } in; in.f = f;
    unsigned r = (in.u + 0x7fffu + ((in.u >> 16) & 1u)) >> 16;  // RNE
    union { unsigned short s; __bf16 b; } out; out.s = (unsigned short)r;
    return out.b;
}

static __device__ __forceinline__ bf16x16 bfzero() {
    bf16x16 z;
#pragma unroll
    for (int i = 0; i < 16; ++i) z[i] = f2bf(0.0f);
    return z;
}

static __device__ __forceinline__ f32x8 wmma_bf16(bf16x16 a, bf16x16 b, f32x8 c) {
    return __builtin_amdgcn_wmma_f32_16x16x32_bf16(false, a, false, b,
                                                   (short)0, c, false, false);
}

// ---------------------------------------------------------------------------
// Weight swizzle: W is (N x K) row-major fp32, used as effective B-matrix
// Beff[k][n] = W[n][k]. Output layout [K/32][nTilesTot][32 lanes][16 elems] bf16
// so a wave loads one B-fragment as a contiguous 32B chunk per lane.
// B-frag (32x16): lane L -> col n = L%16, K-half = L/16, elem i -> k-local i.
// ---------------------------------------------------------------------------
__global__ __launch_bounds__(256)
void swizzle_w(const float* __restrict__ W, __bf16* __restrict__ dst,
               int N, int K, int ntileOff, int nTilesTot) {
    int e = blockIdx.x * 256 + threadIdx.x;
    if (e >= N * K) return;
    int i    = e & 15;
    int lane = (e >> 4) & 31;
    int ntl  = (e >> 9) % (N >> 4);
    int ko   = e / ((N >> 4) << 9);
    int n = ntl * 16 + (lane & 15);
    int k = ko * 32 + (lane >> 4) * 16 + i;
    dst[((size_t)(ko * nTilesTot + ntileOff + ntl) * 32 + lane) * 16 + i] =
        f2bf(W[(size_t)n * K + k]);
}

// ---------------------------------------------------------------------------
// G1: fused projection GEMM. M=32768, K=1024, N=256 (cols 0..127 -> u,
// 128..255 -> sp). WG = 8 waves covering 32 rows x 256 cols; each wave:
// 1 m-subtile x 4 n-tiles, K-loop of 32 chunks, 4 WMMA per chunk.
// ---------------------------------------------------------------------------
__global__ __launch_bounds__(256)
void proj_gemm(const float* __restrict__ x, const __bf16* __restrict__ WiWs_sw,
               const float* __restrict__ bi, const float* __restrict__ bs,
               __bf16* __restrict__ u_bf, float* __restrict__ sp_f) {
    const int lane  = threadIdx.x & 31;
    const int wid   = threadIdx.x >> 5;
    const int msub  = wid >> 2;          // 0..1
    const int ngrp  = wid & 3;           // 0..3 -> n-tiles ngrp*4 .. +3
    const int half  = lane >> 4;
    const int c     = lane & 15;
    const int mrow  = blockIdx.x * 32 + msub * 16 + c;

    f32x8 acc[4];
#pragma unroll
    for (int nt = 0; nt < 4; ++nt) acc[nt] = (f32x8){0, 0, 0, 0, 0, 0, 0, 0};

    for (int ko = 0; ko < D_MODEL / 32; ++ko) {
        // A-fragment of x: lane L holds row (L%16), K = ko*32 + (L/16)*16 + i
        const float4* xp =
            (const float4*)(x + (size_t)mrow * D_MODEL + ko * 32 + half * 16);
        float4 q0 = xp[0], q1 = xp[1], q2 = xp[2], q3 = xp[3];
        bf16x16 af;
        af[0]  = f2bf(q0.x); af[1]  = f2bf(q0.y); af[2]  = f2bf(q0.z); af[3]  = f2bf(q0.w);
        af[4]  = f2bf(q1.x); af[5]  = f2bf(q1.y); af[6]  = f2bf(q1.z); af[7]  = f2bf(q1.w);
        af[8]  = f2bf(q2.x); af[9]  = f2bf(q2.y); af[10] = f2bf(q2.z); af[11] = f2bf(q2.w);
        af[12] = f2bf(q3.x); af[13] = f2bf(q3.y); af[14] = f2bf(q3.z); af[15] = f2bf(q3.w);

        const __bf16* bp =
            WiWs_sw + ((size_t)(ko * 16 + ngrp * 4) * 32 + lane) * 16;
#pragma unroll
        for (int nt = 0; nt < 4; ++nt) {
            bf16x16 bfrag = *(const bf16x16*)(bp + (size_t)nt * 512);
            acc[nt] = wmma_bf16(af, bfrag, acc[nt]);
        }
    }

    // Epilogue: D-frag lane holds col n = L%16, rows r + 8*(L/16).
#pragma unroll
    for (int nt = 0; nt < 4; ++nt) {
        int col = (ngrp * 4 + nt) * 16 + c;
        if (col < D_STATE) {
            float bias = bi[col];
#pragma unroll
            for (int r = 0; r < 8; ++r) {
                int m = blockIdx.x * 32 + msub * 16 + 8 * half + r;
                u_bf[(size_t)m * D_STATE + col] = f2bf(acc[nt][r] + bias);
            }
        } else {
            int cc = col - D_STATE;
            float bias = bs[cc];
#pragma unroll
            for (int r = 0; r < 8; ++r) {
                int m = blockIdx.x * 32 + msub * 16 + 8 * half + r;
                sp_f[(size_t)m * D_STATE + cc] = acc[nt][r] + bias;
            }
        }
    }
}

// ---------------------------------------------------------------------------
// Scan: 1 workgroup, 8 waves. State S = 16x128 bf16 (rows 0..7 = batches,
// 8..15 = zero), stored in LDS in A-fragment layout, ping-pong buffers,
// one barrier per step. Wave w owns output columns [16w, 16w+16).
// Per step: acc = S@A.T + U_t@B.T (8 WMMA), newS = tanh(acc)+sp_t,
// write newS -> LDS, barrier, reload S frags, y_t = S@C.T (4 WMMA).
// ---------------------------------------------------------------------------
__global__ __launch_bounds__(256)
void ssm_scan(const __bf16* __restrict__ u_bf, const float* __restrict__ sp_f,
              const __bf16* __restrict__ A_sw, const __bf16* __restrict__ B_sw,
              const __bf16* __restrict__ C_sw, __bf16* __restrict__ y_bf) {
    __shared__ __align__(32) __bf16 sS[2][2048];  // 2 x (4 chunks x 32 lanes x 16)

    const int lane = threadIdx.x & 31;
    const int w    = threadIdx.x >> 5;  // column-tile owner, 0..7
    const int half = lane >> 4;
    const int c    = lane & 15;

    for (int i = threadIdx.x; i < 2048; i += 256) {
        sS[0][i] = f2bf(0.0f);
        sS[1][i] = f2bf(0.0f);
    }
    __syncthreads();

    // Resident weight B-fragments: A_sw/B_sw/C_sw layout [4][8][32][16].
    bf16x16 Af[4], Bf[4], Cf[4];
#pragma unroll
    for (int kc = 0; kc < 4; ++kc) {
        size_t o = ((size_t)(kc * 8 + w) * 32 + lane) * 16;
        Af[kc] = *(const bf16x16*)(A_sw + o);
        Bf[kc] = *(const bf16x16*)(B_sw + o);
        Cf[kc] = *(const bf16x16*)(C_sw + o);
    }

    bf16x16 Sf[4];
#pragma unroll
    for (int kc = 0; kc < 4; ++kc) Sf[kc] = bfzero();

    const int mb = c;  // batch row for A-fragment lanes
    int cur = 0;

    for (int t = 0; t < SEQ; ++t) {
        f32x8 acc = (f32x8){0, 0, 0, 0, 0, 0, 0, 0};
#pragma unroll
        for (int kc = 0; kc < 4; ++kc) acc = wmma_bf16(Sf[kc], Af[kc], acc);

#pragma unroll
        for (int kc = 0; kc < 4; ++kc) {
            bf16x16 uf;
            if (mb < BATCH)
                uf = *(const bf16x16*)(u_bf + (size_t)(mb * SEQ + t) * D_STATE +
                                       kc * 32 + half * 16);
            else
                uf = bfzero();
            acc = wmma_bf16(uf, Bf[kc], acc);
        }

        // newS = tanh(acc) + sp_t  (D-frag: lane col = 16w+c, rows r+8*half)
        float nv[8];
#pragma unroll
        for (int r = 0; r < 8; ++r) {
            float s = (half == 0)
                          ? sp_f[(size_t)(r * SEQ + t) * D_STATE + w * 16 + c]
                          : 0.0f;
            nv[r] = tanhf(acc[r]) + s;
        }

        int nxt = cur ^ 1;
        if (half == 0) {  // rows 0..7 only; rows 8..15 of S stay zero
#pragma unroll
            for (int r = 0; r < 8; ++r)
                sS[nxt][(w >> 1) * 512 + ((w & 1) * 16 + r) * 16 + c] = f2bf(nv[r]);
        }
        __syncthreads();

#pragma unroll
        for (int kc = 0; kc < 4; ++kc)
            Sf[kc] = *(const bf16x16*)&sS[nxt][kc * 512 + lane * 16];

        f32x8 yacc = (f32x8){0, 0, 0, 0, 0, 0, 0, 0};
#pragma unroll
        for (int kc = 0; kc < 4; ++kc) yacc = wmma_bf16(Sf[kc], Cf[kc], yacc);

        if (half == 0) {
#pragma unroll
            for (int r = 0; r < 8; ++r)
                y_bf[(size_t)(r * SEQ + t) * D_STATE + w * 16 + c] = f2bf(yacc[r]);
        }
        cur = nxt;
    }
}

// ---------------------------------------------------------------------------
// G2: out = y @ Wo.T + bo. M=32768, K=128, N=1024. WG covers 32 rows x 256
// cols; per wave: 4 A-frags (full K), 16 B-frags, 16 WMMA.
// ---------------------------------------------------------------------------
__global__ __launch_bounds__(256)
void out_gemm(const __bf16* __restrict__ y_bf, const __bf16* __restrict__ Wo_sw,
              const float* __restrict__ bo, float* __restrict__ out) {
    const int lane  = threadIdx.x & 31;
    const int wid   = threadIdx.x >> 5;
    const int msub  = wid >> 2;
    const int ngrp  = wid & 3;
    const int half  = lane >> 4;
    const int c     = lane & 15;
    const int mrow  = blockIdx.x * 32 + msub * 16 + c;
    const int ntb   = blockIdx.y * 16 + ngrp * 4;  // base n-tile (of 64)

    bf16x16 afr[4];
#pragma unroll
    for (int kc = 0; kc < 4; ++kc)
        afr[kc] = *(const bf16x16*)(y_bf + (size_t)mrow * D_STATE + kc * 32 +
                                    half * 16);

    f32x8 acc[4];
#pragma unroll
    for (int nt = 0; nt < 4; ++nt) acc[nt] = (f32x8){0, 0, 0, 0, 0, 0, 0, 0};

#pragma unroll
    for (int kc = 0; kc < 4; ++kc) {
#pragma unroll
        for (int nt = 0; nt < 4; ++nt) {
            bf16x16 bfrag = *(const bf16x16*)(
                Wo_sw + ((size_t)(kc * 64 + ntb + nt) * 32 + lane) * 16);
            acc[nt] = wmma_bf16(afr[kc], bfrag, acc[nt]);
        }
    }

#pragma unroll
    for (int nt = 0; nt < 4; ++nt) {
        int col = (ntb + nt) * 16 + c;
        float bias = bo[col];
#pragma unroll
        for (int r = 0; r < 8; ++r) {
            int m = blockIdx.x * 32 + msub * 16 + 8 * half + r;
            out[(size_t)m * D_MODEL + col] = acc[nt][r] + bias;
        }
    }
}

// ---------------------------------------------------------------------------
extern "C" void kernel_launch(void* const* d_in, const int* in_sizes, int n_in,
                              void* d_out, int out_size, void* d_ws, size_t ws_size,
                              hipStream_t stream) {
    (void)in_sizes; (void)n_in; (void)out_size; (void)ws_size;

    const float* x  = (const float*)d_in[0];
    const float* Wi = (const float*)d_in[1];
    const float* bi = (const float*)d_in[2];
    const float* Ws = (const float*)d_in[3];
    const float* bs = (const float*)d_in[4];
    const float* Wo = (const float*)d_in[5];
    const float* bo = (const float*)d_in[6];
    const float* A  = (const float*)d_in[7];
    const float* Bm = (const float*)d_in[8];
    const float* C  = (const float*)d_in[9];

    char* ws = (char*)d_ws;
    size_t off = 0;
    auto alloc = [&](size_t bytes) -> void* {
        off = (off + 255) & ~(size_t)255;
        void* p = ws + off;
        off += bytes;
        return p;
    };

    __bf16* WiWs_sw = (__bf16*)alloc((size_t)32 * 16 * 512 * 2);        // 512 KB
    __bf16* Wo_sw   = (__bf16*)alloc((size_t)4 * 64 * 512 * 2);         // 256 KB
    __bf16* A_sw    = (__bf16*)alloc((size_t)4 * 8 * 512 * 2);
    __bf16* B_sw    = (__bf16*)alloc((size_t)4 * 8 * 512 * 2);
    __bf16* C_sw    = (__bf16*)alloc((size_t)4 * 8 * 512 * 2);
    __bf16* u_bf    = (__bf16*)alloc((size_t)M_TOTAL * D_STATE * 2);    // 8 MB
    float*  sp_f    = (float*)alloc((size_t)M_TOTAL * D_STATE * 4);     // 16 MB
    __bf16* y_bf    = (__bf16*)alloc((size_t)M_TOTAL * D_STATE * 2);    // 8 MB

    // Phase P: weight conversion + swizzle into WMMA B-fragment layouts.
    swizzle_w<<<512, 256, 0, stream>>>(Wi, WiWs_sw, 128, 1024, 0, 16);
    swizzle_w<<<512, 256, 0, stream>>>(Ws, WiWs_sw, 128, 1024, 8, 16);
    swizzle_w<<<512, 256, 0, stream>>>(Wo, Wo_sw, 1024, 128, 0, 64);
    swizzle_w<<<64, 256, 0, stream>>>(A, A_sw, 128, 128, 0, 8);
    swizzle_w<<<64, 256, 0, stream>>>(Bm, B_sw, 128, 128, 0, 8);
    swizzle_w<<<64, 256, 0, stream>>>(C, C_sw, 128, 128, 0, 8);

    // Phase G1: fused u/sp projections.
    proj_gemm<<<M_TOTAL / 32, 256, 0, stream>>>(x, WiWs_sw, bi, bs, u_bf, sp_f);

    // Phase S: sequential scan (latency-bound, single WG of 8 waves).
    ssm_scan<<<1, 256, 0, stream>>>(u_bf, sp_f, A_sw, B_sw, C_sw, y_bf);

    // Phase G2: output projection.
    out_gemm<<<dim3(M_TOTAL / 32, 4), 256, 0, stream>>>(y_bf, Wo_sw, bo,
                                                        (float*)d_out);
}
